// CGCAM_39324720562626
// MI455X (gfx1250) — matrix operand order repaired and checked by reading, not compiled, not run on hardware
//
#include <hip/hip_runtime.h>

// ---------------------------------------------------------------------------
// Problem constants
// ---------------------------------------------------------------------------
#define Bb 8
#define Cc 512
#define Hh 96
#define Ww 96
#define NH 8
#define DH 64
#define Nn (Bb * Ww)        // 768 attention "batches"
#define Mrows (Nn * Hh)     // 73728 GEMM M
#define ELEMS ((size_t)Bb * Cc * Hh * Ww)   // 37748736
#define WSZ (512 * 512)

typedef __bf16 bf16;
typedef __attribute__((ext_vector_type(16))) __bf16 v16bf;
typedef __attribute__((ext_vector_type(8)))  __bf16 v8bf;
typedef __attribute__((ext_vector_type(8)))  float  v8f;

union Frag {
    v16bf v;
    v8bf  h[2];
    bf16  e[16];
};

__device__ __forceinline__ v8f wmma_bf16(const Frag& a, const Frag& b, v8f c) {
    return __builtin_amdgcn_wmma_f32_16x16x32_bf16(
        false, a.v, false, b.v, (short)0, c, false, false);
}

// ---------------------------------------------------------------------------
// Weight conversion f32 -> bf16 (7 x 512x512 matrices packed back-to-back)
// ---------------------------------------------------------------------------
__global__ __launch_bounds__(256) void k_cvt_weights(
    const float* __restrict__ w0, const float* __restrict__ w1,
    const float* __restrict__ w2, const float* __restrict__ w3,
    const float* __restrict__ w4, const float* __restrict__ w5,
    const float* __restrict__ w6, bf16* __restrict__ out)
{
    unsigned i = blockIdx.x * 256u + threadIdx.x;
    if (i >= 7u * WSZ) return;
    unsigned which = i >> 18;
    unsigned j = i & (WSZ - 1u);
    const float* src;
    switch (which) {
        case 0: src = w0; break;
        case 1: src = w1; break;
        case 2: src = w2; break;
        case 3: src = w3; break;
        case 4: src = w4; break;
        case 5: src = w5; break;
        default: src = w6; break;
    }
    out[i] = (bf16)src[j];
}

// ---------------------------------------------------------------------------
// Build vqk[(b*W+w), h, c] (bf16) from x[b,c,h,w] (f32)
// ---------------------------------------------------------------------------
__global__ __launch_bounds__(256) void k_build_vqk(
    const float* __restrict__ x, bf16* __restrict__ vqk)
{
    size_t f = (size_t)blockIdx.x * 256u + threadIdx.x;
    if (f >= ELEMS) return;
    unsigned c = (unsigned)(f & 511u);
    unsigned h = (unsigned)((f >> 9) % 96u);
    unsigned w = (unsigned)((f / 49152u) % 96u);
    unsigned b = (unsigned)(f / 4718592u);
    vqk[f] = (bf16)x[(((size_t)b * 512u + c) * 96u + h) * 96u + w];
}

// ---------------------------------------------------------------------------
// q_view permutation:  qv[f] = o[((b*96+w)*96+h)*512 + c]
// ---------------------------------------------------------------------------
__global__ __launch_bounds__(256) void k_qview(
    const bf16* __restrict__ o, bf16* __restrict__ qv)
{
    size_t f = (size_t)blockIdx.x * 256u + threadIdx.x;
    if (f >= ELEMS) return;
    unsigned w = (unsigned)(f % 96u);
    unsigned h = (unsigned)((f / 96u) % 96u);
    unsigned c = (unsigned)((f / 9216u) % 512u);
    unsigned b = (unsigned)(f / 4718592u);
    qv[f] = o[((size_t)(b * 96u + w) * 96u + h) * 512u + c];
}

// ---------------------------------------------------------------------------
// Depthwise 3x3 (SAME) + BN(eval) -> A_dw[(b,h,w), c]  (bf16, channels-last)
// ---------------------------------------------------------------------------
__global__ __launch_bounds__(256) void k_dwconv_bn(
    const float* __restrict__ x, const float* __restrict__ dww,
    const float* __restrict__ bng, const float* __restrict__ bnb,
    const float* __restrict__ bnm, const float* __restrict__ bnv,
    bf16* __restrict__ adw)
{
    size_t idx = (size_t)blockIdx.x * 256u + threadIdx.x;
    if (idx >= ELEMS) return;
    unsigned w = (unsigned)(idx % 96u);
    unsigned h = (unsigned)((idx / 96u) % 96u);
    unsigned c = (unsigned)((idx / 9216u) % 512u);
    unsigned b = (unsigned)(idx / 4718592u);

    const float* xp = x + ((size_t)b * 512u + c) * 9216u;
    const float* wp = dww + c * 9u;
    float s = 0.f;
#pragma unroll
    for (int dy = -1; dy <= 1; ++dy) {
#pragma unroll
        for (int dx = -1; dx <= 1; ++dx) {
            int yy = (int)h + dy, xx = (int)w + dx;
            if (yy >= 0 && yy < 96 && xx >= 0 && xx < 96)
                s += xp[yy * 96 + xx] * wp[(dy + 1) * 3 + (dx + 1)];
        }
    }
    float sc = bng[c] * rsqrtf(bnv[c] + 1e-5f);
    float v = (s - bnm[c]) * sc + bnb[c];
    adw[((size_t)(b * 96u + h) * 96u + w) * 512u + c] = (bf16)v;
}

// ---------------------------------------------------------------------------
// K=512, N=512 GEMM:  out = A[M,512] @ W.T   (W is [N,K] row-major)
// 128 threads = 4 waves in a 2(M)x2(N) arrangement.
// Per wave: 32(M) x 64(N) register tile = 2 A-frags x 4 B-frags, 8 WMMA/K-step.
// Block tile: 64(M) x 128(N).  grid = (1152, 4).
// mode 0: bf16 out (+bias, +residual)   mode 1: f32 out
// mode 2: clip(acc,0,6) + cr0 -> NCHW f32
// ---------------------------------------------------------------------------
__global__ __launch_bounds__(128) void k_gemm512(
    const bf16* A, const bf16* __restrict__ W,
    const float* bias, const bf16* resid,
    bf16* outB, float* outF,
    const float* cr0, float* outNCHW, int mode)
{
    const int lane  = threadIdx.x & 31;
    const int wave  = threadIdx.x >> 5;
    const int waveM = wave >> 1;
    const int waveN = wave & 1;
    const int m0 = blockIdx.x * 64 + waveM * 32;
    const int n0 = blockIdx.y * 128 + waveN * 64;
    const int mr   = lane & 15;
    const int ksel = lane >> 4;

    const bf16* arow0 = A + (size_t)(m0 + mr) * 512u;
    const bf16* arow1 = A + (size_t)(m0 + 16 + mr) * 512u;
    const bf16* wrow0 = W + (size_t)(n0 + mr) * 512u;

    v8f acc[2][4];
#pragma unroll
    for (int ai = 0; ai < 2; ++ai)
#pragma unroll
        for (int bi = 0; bi < 4; ++bi) acc[ai][bi] = (v8f){};

#pragma unroll 2
    for (int k0 = 0; k0 < 512; k0 += 32) {
        Frag a[2], b[4];
        a[0].h[0] = *(const v8bf*)(arow0 + k0 + 16 * ksel);
        a[0].h[1] = *(const v8bf*)(arow0 + k0 + 8 + 16 * ksel);
        a[1].h[0] = *(const v8bf*)(arow1 + k0 + 16 * ksel);
        a[1].h[1] = *(const v8bf*)(arow1 + k0 + 8 + 16 * ksel);
#pragma unroll
        for (int bi = 0; bi < 4; ++bi)
            b[bi].v = *(const v16bf*)(wrow0 + (size_t)bi * 16u * 512u + k0 + 16 * ksel);
#pragma unroll
        for (int ai = 0; ai < 2; ++ai)
#pragma unroll
            for (int bi = 0; bi < 4; ++bi)
                acc[ai][bi] = wmma_bf16(a[ai], b[bi], acc[ai][bi]);
    }

    if (mode == 0) {
#pragma unroll
        for (int ai = 0; ai < 2; ++ai)
#pragma unroll
            for (int bi = 0; bi < 4; ++bi) {
                const int dn = n0 + 16 * bi + mr;
                const float bv = bias ? bias[dn] : 0.f;
#pragma unroll
                for (int r = 0; r < 8; ++r) {
                    int dm = m0 + 16 * ai + 8 * ksel + r;
                    size_t off = (size_t)dm * 512u + dn;
                    float v = acc[ai][bi][r] + bv;
                    if (resid) v += (float)resid[off];
                    outB[off] = (bf16)v;
                }
            }
    } else if (mode == 1) {
#pragma unroll
        for (int ai = 0; ai < 2; ++ai)
#pragma unroll
            for (int bi = 0; bi < 4; ++bi) {
                const int dn = n0 + 16 * bi + mr;
#pragma unroll
                for (int r = 0; r < 8; ++r) {
                    int dm = m0 + 16 * ai + 8 * ksel + r;
                    outF[(size_t)dm * 512u + dn] = acc[ai][bi][r];
                }
            }
    } else {
#pragma unroll
        for (int ai = 0; ai < 2; ++ai)
#pragma unroll
            for (int bi = 0; bi < 4; ++bi) {
                const int dn = n0 + 16 * bi + mr;
#pragma unroll
                for (int r = 0; r < 8; ++r) {
                    int p = m0 + 16 * ai + 8 * ksel + r;      // (b,h,w) position
                    int bb = p / 9216;
                    int rem = p - bb * 9216;
                    int hh = rem / 96;
                    int wp = rem - hh * 96;
                    float v = fminf(fmaxf(acc[ai][bi][r], 0.f), 6.f);
                    v += cr0[((size_t)(bb * 96 + wp) * 96u + hh) * 512u + dn];
                    outNCHW[(((size_t)bb * 512u + dn) * 96u + hh) * 96u + wp] = v;
                }
            }
    }
}

// ---------------------------------------------------------------------------
// Attention for one (batch n, head h).  LDS layout (dynamic, base offset 0):
//   [0      ) sQ  96x64 bf16   (12288 B)   <- async copy
//   [12288  ) sK  96x64 bf16   (12288 B)   <- async copy
//   [24576  ) sVt 64x96 bf16   (12288 B)   <- transposed so P@V B-frags are
//                                             contiguous v16bf LDS loads
//   [36864  ) sS  96x96 f32    (36864 B)
//   [73728  ) sP  96x96 bf16   (18432 B)
// ---------------------------------------------------------------------------
__global__ __launch_bounds__(192) void k_attn96(
    const bf16* __restrict__ Q, const bf16* __restrict__ K,
    const bf16* __restrict__ V, bf16* __restrict__ O)
{
    const int pair = blockIdx.x;
    const int bn = pair >> 3;
    const int hd = pair & 7;

    extern __shared__ char smem[];
    bf16*  sQ  = (bf16*)smem;
    bf16*  sK  = sQ + 96 * 64;
    bf16*  sVt = sK + 96 * 64;                // [64][96]
    float* sS  = (float*)(sVt + 64 * 96);
    bf16*  sP  = (bf16*)(sS + 96 * 96);

    const size_t base = ((size_t)bn * 96u) * 512u + (size_t)hd * 64u;
    const bf16* qg = Q + base;
    const bf16* kg = K + base;
    const bf16* vg = V + base;

    // ---- async copy Q,K tiles into LDS (16B per lane per op, ASYNCcnt) ----
    for (int i = threadIdx.x; i < 96 * 8; i += 192) {
        int t = i >> 3, d8 = (i & 7) * 8;
        unsigned qoff = (unsigned)((t * 64 + d8) * 2);          // sQ at LDS 0
        unsigned koff = 12288u + qoff;                          // sK
        unsigned long long qa = (unsigned long long)(qg + (size_t)t * 512u + d8);
        unsigned long long ka = (unsigned long long)(kg + (size_t)t * 512u + d8);
        asm volatile("global_load_async_to_lds_b128 %0, %1, off"
                     :: "v"(qoff), "v"(qa) : "memory");
        asm volatile("global_load_async_to_lds_b128 %0, %1, off"
                     :: "v"(koff), "v"(ka) : "memory");
    }
    // ---- V transposed into LDS (coalesced global reads) ----
    for (int i = threadIdx.x; i < 96 * 64; i += 192) {
        int d = i & 63, t = i >> 6;
        sVt[d * 96 + t] = vg[(size_t)t * 512u + d];
    }
    asm volatile("s_wait_asynccnt 0x0" ::: "memory");
    __syncthreads();

    const int lane = threadIdx.x & 31;
    const int wave = threadIdx.x >> 5;   // 0..5
    const int m0 = wave * 16;
    const int mr = lane & 15;
    const int ksel = lane >> 4;

    // ---- scores: S = Q K^T * 0.125 ----
    {
        v8f s[6];
#pragma unroll
        for (int nt = 0; nt < 6; ++nt) s[nt] = (v8f){};
#pragma unroll
        for (int k0 = 0; k0 < 64; k0 += 32) {
            Frag a;
            a.h[0] = *(const v8bf*)(sQ + (m0 + mr) * 64 + k0 + 16 * ksel);
            a.h[1] = *(const v8bf*)(sQ + (m0 + mr) * 64 + k0 + 8 + 16 * ksel);
#pragma unroll
            for (int nt = 0; nt < 6; ++nt) {
                Frag b;
                b.v = *(const v16bf*)(sK + (nt * 16 + mr) * 64 + k0 + 16 * ksel);
                s[nt] = wmma_bf16(a, b, s[nt]);
            }
        }
#pragma unroll
        for (int nt = 0; nt < 6; ++nt)
#pragma unroll
            for (int r = 0; r < 8; ++r)
                sS[(m0 + 8 * ksel + r) * 96 + nt * 16 + mr] = s[nt][r] * 0.125f;
    }
    __syncthreads();

    // ---- softmax over rows ----
    for (int row = threadIdx.x; row < 96; row += 192) {
        float* rp = sS + row * 96;
        float mx = -3.0e38f;
        for (int j = 0; j < 96; ++j) mx = fmaxf(mx, rp[j]);
        float sum = 0.f;
        for (int j = 0; j < 96; ++j) { float e = __expf(rp[j] - mx); rp[j] = e; sum += e; }
        float inv = 1.f / sum;
        bf16* pp = sP + row * 96;
        for (int j = 0; j < 96; ++j) pp[j] = (bf16)(rp[j] * inv);
    }
    __syncthreads();

    // ---- O = P @ V  (B-frags contiguous from sVt) ----
    {
        v8f o[4];
#pragma unroll
        for (int nt = 0; nt < 4; ++nt) o[nt] = (v8f){};
#pragma unroll
        for (int k0 = 0; k0 < 96; k0 += 32) {
            Frag a;
            a.h[0] = *(const v8bf*)(sP + (m0 + mr) * 96 + k0 + 16 * ksel);
            a.h[1] = *(const v8bf*)(sP + (m0 + mr) * 96 + k0 + 8 + 16 * ksel);
            const int kb = k0 + 16 * ksel;
#pragma unroll
            for (int nt = 0; nt < 4; ++nt) {
                Frag b;
                b.v = *(const v16bf*)(sVt + (nt * 16 + mr) * 96 + kb);
                o[nt] = wmma_bf16(a, b, o[nt]);
            }
        }
        bf16* og = O + base;
#pragma unroll
        for (int nt = 0; nt < 4; ++nt)
#pragma unroll
            for (int r = 0; r < 8; ++r) {
                int t = m0 + 8 * ksel + r;
                og[(size_t)t * 512u + nt * 16 + mr] = (bf16)o[nt][r];
            }
    }
}

// ---------------------------------------------------------------------------
// Host-side orchestration
// ---------------------------------------------------------------------------
extern "C" void kernel_launch(void* const* d_in, const int* in_sizes, int n_in,
                              void* d_out, int out_size, void* d_ws, size_t ws_size,
                              hipStream_t stream)
{
    const float* x      = (const float*)d_in[0];
    const float* msa_Wk = (const float*)d_in[1];
    const float* msa_Wv = (const float*)d_in[2];
    const float* msa_Wo = (const float*)d_in[3];
    const float* cr_Wq  = (const float*)d_in[4];
    const float* cr_bq  = (const float*)d_in[5];
    const float* cr_Wv  = (const float*)d_in[6];
    const float* cr_Wo  = (const float*)d_in[7];
    const float* dw_w   = (const float*)d_in[8];
    const float* bn_g   = (const float*)d_in[9];
    const float* bn_b   = (const float*)d_in[10];
    const float* bn_m   = (const float*)d_in[11];
    const float* bn_v   = (const float*)d_in[12];
    const float* pw_w   = (const float*)d_in[13];
    float* out = (float*)d_out;

    bf16* ws = (bf16*)d_ws;
    const size_t SZ = ELEMS;
    bf16* R0 = ws;
    bf16* R1 = ws + SZ;
    bf16* R2 = ws + 2 * SZ;
    bf16* R3 = ws + 3 * SZ;
    bf16* Wts = ws + 4 * SZ;
    bf16* Wk  = Wts + 0 * WSZ;
    bf16* Wv  = Wts + 1 * WSZ;
    bf16* Wo  = Wts + 2 * WSZ;
    bf16* cWq = Wts + 3 * WSZ;
    bf16* cWv = Wts + 4 * WSZ;
    bf16* cWo = Wts + 5 * WSZ;
    bf16* pw  = Wts + 6 * WSZ;
    float* cr0 = (float*)R0;                 // overlays R0+R1 when in use

    const int EB = (int)((ELEMS + 255) / 256);
    dim3 gg(Mrows / 64, 512 / 128);          // (1152, 4)
    const size_t attn_lds = (size_t)(3 * 96 * 64 * 2 + 96 * 96 * 4 + 96 * 96 * 2);

    k_cvt_weights<<<(7 * WSZ + 255) / 256, 256, 0, stream>>>(
        msa_Wk, msa_Wv, msa_Wo, cr_Wq, cr_Wv, cr_Wo, pw_w, Wts);
    k_build_vqk<<<EB, 256, 0, stream>>>(x, R0);

    // MSA
    k_gemm512<<<gg, 128, 0, stream>>>(R0, Wk, nullptr, R0, R1, nullptr, nullptr, nullptr, 0);
    k_gemm512<<<gg, 128, 0, stream>>>(R0, Wv, nullptr, nullptr, R2, nullptr, nullptr, nullptr, 0);
    k_attn96<<<Nn * NH, 192, attn_lds, stream>>>(R0, R1, R2, R3);
    k_gemm512<<<gg, 128, 0, stream>>>(R3, Wo, nullptr, nullptr, R1, nullptr, nullptr, nullptr, 0);
    k_qview<<<EB, 256, 0, stream>>>(R1, R2);

    // Cross-attention
    k_gemm512<<<gg, 128, 0, stream>>>(R2, cWq, cr_bq, nullptr, R3, nullptr, nullptr, nullptr, 0);
    k_gemm512<<<gg, 128, 0, stream>>>(R0, cWv, nullptr, nullptr, R1, nullptr, nullptr, nullptr, 0);
    k_attn96<<<Nn * NH, 192, attn_lds, stream>>>(R3, R1, R1, R2);
    k_gemm512<<<gg, 128, 0, stream>>>(R2, cWo, nullptr, nullptr, nullptr, cr0, nullptr, nullptr, 1);

    // Conv branch + fused final epilogue
    k_dwconv_bn<<<EB, 256, 0, stream>>>(x, dw_w, bn_g, bn_b, bn_m, bn_v, R3);
    k_gemm512<<<gg, 128, 0, stream>>>(R3, pw, nullptr, nullptr, nullptr, nullptr, cr0, out, 2);
}
